// MusicTransformerAttention_36593121362539
// MI455X (gfx1250) — matrix operand order repaired and checked by reading, not compiled, not run on hardware
//
#include <hip/hip_runtime.h>
#include <hip/hip_bf16.h>

// ---------------------------------------------------------------------------
// MusicTransformer attention for MI455X (gfx1250, wave32, WMMA).
// B=4, S=2048, HID=512, NH=8, HD=64, MAX_REL=512.
// bf16 WMMA (16x16x32, f32 accum) for all matmuls; flash-attention streaming
// with online softmax; global->LDS staging via gfx1250 async-LDS loads.
// ---------------------------------------------------------------------------

typedef __bf16 bf16;
typedef __attribute__((ext_vector_type(8)))  bf16  v8bf;
typedef __attribute__((ext_vector_type(16))) bf16  v16bf;
typedef __attribute__((ext_vector_type(8)))  float v8f;

static constexpr int BB  = 4;
static constexpr int SS  = 2048;
static constexpr int HID = 512;
static constexpr int NH  = 8;
static constexpr int HD  = 64;

__device__ __forceinline__ v16bf cat16(v8bf lo, v8bf hi) {
  return __builtin_shufflevector(lo, hi, 0,1,2,3,4,5,6,7,8,9,10,11,12,13,14,15);
}

__device__ __forceinline__ v8f wmma_bf16(v16bf a, v16bf b, v8f c) {
  return __builtin_amdgcn_wmma_f32_16x16x32_bf16(
      false, a, false, b, (short)0, c, false, false);
}

// ---------------------------------------------------------------------------
// gfx1250 async global->LDS 16-byte copy (ASYNCcnt-tracked data mover path).
// ---------------------------------------------------------------------------
#define USE_ASYNC_LDS 1

typedef __attribute__((address_space(3))) void lds_void_t;

__device__ __forceinline__ unsigned lds_byte_off(void* p) {
  return (unsigned)(uintptr_t)(lds_void_t*)p;
}

__device__ __forceinline__ void copy16B_g2l(bf16* ldst, const bf16* gsrc) {
#if USE_ASYNC_LDS
  unsigned           loff = lds_byte_off(ldst);
  unsigned long long gaddr = (unsigned long long)(uintptr_t)gsrc;
  asm volatile("global_load_async_to_lds_b128 %0, %1, off"
               :: "v"(loff), "v"(gaddr) : "memory");
#else
  *(v8bf*)ldst = *(const v8bf*)gsrc;
#endif
}

__device__ __forceinline__ void async_fence() {
#if USE_ASYNC_LDS
  asm volatile("s_wait_asynccnt 0" ::: "memory");
#endif
}

// ---------------------------------------------------------------------------
// f32 -> bf16 conversion (hidden states)
// ---------------------------------------------------------------------------
__global__ void cvt_bf16_kernel(const float* __restrict__ in,
                                bf16* __restrict__ out, int n) {
  int i = blockIdx.x * blockDim.x + threadIdx.x;
  if (i < n) out[i] = (bf16)in[i];
}

// ---------------------------------------------------------------------------
// f32 [512][512] -> bf16 transposed [n][k] (weights), tiled via LDS.
// grid (16,16), 256 threads (32x8).
// ---------------------------------------------------------------------------
__global__ __launch_bounds__(256) void cvt_bf16_transpose_kernel(
    const float* __restrict__ in, bf16* __restrict__ out) {
  __shared__ float tile[32][33];
  const int bx = blockIdx.x * 32;  // k offset
  const int by = blockIdx.y * 32;  // n offset
  const int tx = threadIdx.x & 31;
  const int ty = threadIdx.x >> 5;
#pragma unroll
  for (int r = 0; r < 32; r += 8)
    tile[ty + r][tx] = in[(size_t)(bx + ty + r) * HID + by + tx];
  __syncthreads();
#pragma unroll
  for (int r = 0; r < 32; r += 8)
    out[(size_t)(by + ty + r) * HID + bx + tx] = (bf16)tile[tx][ty + r];
}

// ---------------------------------------------------------------------------
// GEMM: C[8192][512] = A[8192][512](bf16) * Wt^T + bias, Wt pre-transposed
// [n][k]. Block tile 128x64, 8 waves (4x2), each wave 32x32 (2x2 WMMA).
// MODE 0: f32 row-major out.  MODE 1: bf16 [bh][s][d] scaled.  MODE 2: bf16
// [bh][d][s] (V transposed).
// ---------------------------------------------------------------------------
template <int MODE>
__global__ __launch_bounds__(256) void gemm512(const bf16* __restrict__ A,
                                               const bf16* __restrict__ Wt,
                                               const float* __restrict__ bias,
                                               void* __restrict__ out,
                                               float scale) {
  __shared__ bf16 As[128 * 40];   // [row][k], k-stride 40
  __shared__ bf16 Bs[64 * 40];    // [n_local][k]

  const int tid  = threadIdx.x;
  const int lane = tid & 31;
  const int wv   = tid >> 5;
  const int wm   = wv >> 1;
  const int wn   = wv & 1;
  const int m0   = blockIdx.x * 128;
  const int n0   = blockIdx.y * 64;
  const int lrow = lane & 15;
  const int hi   = lane >> 4;

  v8f acc[2][2] = {};

  for (int kt = 0; kt < HID; kt += 32) {
    {  // A tile: 128x32, two 16B async copies per thread.
      int row = tid >> 1, hh = tid & 1;
      const bf16* src = A + (size_t)(m0 + row) * HID + kt + hh * 16;
      copy16B_g2l(&As[row * 40 + hh * 16],     src);
      copy16B_g2l(&As[row * 40 + hh * 16 + 8], src + 8);
    }
    {  // B tile: 64(n) x 32(k) from pre-transposed W, one copy per thread.
      int n = tid >> 2, kc = (tid & 3) * 8;
      copy16B_g2l(&Bs[n * 40 + kc], Wt + (size_t)(n0 + n) * HID + kt + kc);
    }
    async_fence();
    __syncthreads();

    v16bf af[2], bfr[2];
#pragma unroll
    for (int mi = 0; mi < 2; ++mi) {
      int m = wm * 32 + mi * 16 + lrow;
      af[mi] = cat16(*(const v8bf*)&As[m * 40 + hi * 8],
                     *(const v8bf*)&As[m * 40 + 16 + hi * 8]);
    }
#pragma unroll
    for (int ni = 0; ni < 2; ++ni) {
      int n = wn * 32 + ni * 16 + lrow;
      bfr[ni] = cat16(*(const v8bf*)&Bs[n * 40 + hi * 8],
                      *(const v8bf*)&Bs[n * 40 + 16 + hi * 8]);
    }
#pragma unroll
    for (int mi = 0; mi < 2; ++mi)
#pragma unroll
      for (int ni = 0; ni < 2; ++ni)
        acc[mi][ni] = wmma_bf16(af[mi], bfr[ni], acc[mi][ni]);
    __syncthreads();
  }

#pragma unroll
  for (int mi = 0; mi < 2; ++mi)
#pragma unroll
    for (int ni = 0; ni < 2; ++ni) {
      int col = n0 + wn * 32 + ni * 16 + lrow;
      float bv = bias[col];
#pragma unroll
      for (int e = 0; e < 8; ++e) {
        int row = m0 + wm * 32 + mi * 16 + hi * 8 + e;
        float v = (acc[mi][ni][e] + bv) * scale;
        if (MODE == 0) {
          ((float*)out)[(size_t)row * HID + col] = v;
        } else {
          int b = row >> 11, s = row & 2047;
          int h = col >> 6,  d = col & 63;
          size_t bh = (size_t)(b * NH + h);
          if (MODE == 1)
            ((bf16*)out)[(bh * SS + s) * HD + d] = (bf16)v;
          else
            ((bf16*)out)[(bh * HD + d) * SS + s] = (bf16)v;
        }
      }
    }
}

// ---------------------------------------------------------------------------
// Flash attention. grid = (S/128, B*NH). 256 threads = 8 waves; each wave
// owns 16 query rows. 64-key blocks staged via async-LDS; 16 WMMAs/block.
// Bias via pre-clamped padded table: relPad[t] = rel_table[clamp(t-2048)][h],
// so bias(i,j) = relPad[j-i+2048] with no clamp math in the hot loop.
// ---------------------------------------------------------------------------
__global__ __launch_bounds__(256) void attn_kernel(const bf16* __restrict__ Q,
                                                   const bf16* __restrict__ K,
                                                   const bf16* __restrict__ Vt,
                                                   const float* __restrict__ rel_table,
                                                   const float* __restrict__ mask,
                                                   bf16* __restrict__ Ctx) {
  __shared__ bf16  Ks[64 * 72];       // [key_local][d]
  __shared__ bf16  Vs[64 * 72];       // [d][key_local]
  __shared__ bf16  Ps[8 * 16 * 72];   // per-wave probs
  __shared__ float relPad[4096];      // padded pre-clamped bias column
  __shared__ float maskAdd[64];

  const int tid  = threadIdx.x;
  const int lane = tid & 31;
  const int wv   = tid >> 5;
  const int bh   = blockIdx.y;
  const int b    = bh >> 3;
  const int h    = bh & 7;
  const int q0   = blockIdx.x * 128;
  const int lrow = lane & 15;
  const int hi   = lane >> 4;

  for (int i = tid; i < 4096; i += 256) {
    int rel = i - 2048;
    rel = rel < -512 ? -512 : (rel > 512 ? 512 : rel);
    relPad[i] = rel_table[(rel + 512) * NH + h];
  }

  const bf16* Qh = Q  + (size_t)bh * SS * HD;
  const bf16* Kh = K  + (size_t)bh * SS * HD;
  const bf16* Vh = Vt + (size_t)bh * HD * SS;

  v16bf qf[2];
  {
    const bf16* qr = Qh + (size_t)(q0 + wv * 16 + lrow) * HD;
#pragma unroll
    for (int f = 0; f < 2; ++f)
      qf[f] = cat16(*(const v8bf*)(qr + f * 32 + hi * 8),
                    *(const v8bf*)(qr + f * 32 + 16 + hi * 8));
  }

  float mrow[8], lrun[8];
#pragma unroll
  for (int e = 0; e < 8; ++e) { mrow[e] = -3.0e38f; lrun[e] = 0.0f; }
  v8f accc[4] = {};

  const int ibase = q0 + wv * 16 + hi * 8;   // row of element e=0

  for (int kb = 0; kb < SS; kb += 64) {
    {  // K and V tiles: 64x64 bf16 each, 4 async 16B copies per thread.
      int r = tid >> 2, qd = (tid & 3) * 16;
      const bf16* ksrc = Kh + (size_t)(kb + r) * HD + qd;
      copy16B_g2l(&Ks[r * 72 + qd],     ksrc);
      copy16B_g2l(&Ks[r * 72 + qd + 8], ksrc + 8);
      const bf16* vsrc = Vh + (size_t)r * SS + kb + qd;
      copy16B_g2l(&Vs[r * 72 + qd],     vsrc);
      copy16B_g2l(&Vs[r * 72 + qd + 8], vsrc + 8);
    }
    if (tid < 64) {
      float mk = mask[b * SS + kb + tid];
      maskAdd[tid] = (1.0f - mk) * -3.3895314e38f;
    }
    async_fence();
    __syncthreads();

    // Scores S = Q*K^T (8 WMMAs per wave) + bias + mask.
    v8f sc[4];
    const int tbase = kb + lrow - ibase + 2048;  // relPad index for nt=0,e=0
#pragma unroll
    for (int nt = 0; nt < 4; ++nt) {
      int key = nt * 16 + lrow;
      v16bf kf0 = cat16(*(const v8bf*)&Ks[key * 72 + hi * 8],
                        *(const v8bf*)&Ks[key * 72 + 16 + hi * 8]);
      v16bf kf1 = cat16(*(const v8bf*)&Ks[key * 72 + 32 + hi * 8],
                        *(const v8bf*)&Ks[key * 72 + 32 + 16 + hi * 8]);
      v8f z = {};
      sc[nt] = wmma_bf16(qf[0], kf0, z);
      sc[nt] = wmma_bf16(qf[1], kf1, sc[nt]);

      float madd = maskAdd[nt * 16 + lrow];
#pragma unroll
      for (int e = 0; e < 8; ++e)
        sc[nt][e] += relPad[tbase + nt * 16 - e] + madd;
    }

    // Online softmax.
#pragma unroll
    for (int e = 0; e < 8; ++e) {
      float vmax = fmaxf(fmaxf(sc[0][e], sc[1][e]), fmaxf(sc[2][e], sc[3][e]));
#pragma unroll
      for (int off = 8; off >= 1; off >>= 1)
        vmax = fmaxf(vmax, __shfl_xor(vmax, off, 32));
      float mnew = fmaxf(mrow[e], vmax);
      float corr = __builtin_amdgcn_exp2f((mrow[e] - mnew) * 1.44269504f);
      float psum = 0.0f;
#pragma unroll
      for (int nt = 0; nt < 4; ++nt) {
        float p = __builtin_amdgcn_exp2f((sc[nt][e] - mnew) * 1.44269504f);
        sc[nt][e] = p;
        psum += p;
      }
#pragma unroll
      for (int off = 8; off >= 1; off >>= 1)
        psum += __shfl_xor(psum, off, 32);
      lrun[e] = lrun[e] * corr + psum;
      mrow[e] = mnew;
#pragma unroll
      for (int nt = 0; nt < 4; ++nt) accc[nt][e] *= corr;
    }

    // Stage P (bf16) to this wave's slab; re-read in A-fragment layout.
    bf16* pw = &Ps[wv * 16 * 72];
#pragma unroll
    for (int nt = 0; nt < 4; ++nt)
#pragma unroll
      for (int e = 0; e < 8; ++e)
        pw[(hi * 8 + e) * 72 + nt * 16 + lrow] = (bf16)sc[nt][e];

    v16bf pf[2];
#pragma unroll
    for (int f = 0; f < 2; ++f)
      pf[f] = cat16(*(const v8bf*)&pw[lrow * 72 + f * 32 + hi * 8],
                    *(const v8bf*)&pw[lrow * 72 + f * 32 + 16 + hi * 8]);

    // ctx += P*V (8 WMMAs per wave).
#pragma unroll
    for (int nt = 0; nt < 4; ++nt) {
      int d = nt * 16 + lrow;
      v16bf vf0 = cat16(*(const v8bf*)&Vs[d * 72 + hi * 8],
                        *(const v8bf*)&Vs[d * 72 + 16 + hi * 8]);
      v16bf vf1 = cat16(*(const v8bf*)&Vs[d * 72 + 32 + hi * 8],
                        *(const v8bf*)&Vs[d * 72 + 32 + 16 + hi * 8]);
      accc[nt] = wmma_bf16(pf[0], vf0, accc[nt]);
      accc[nt] = wmma_bf16(pf[1], vf1, accc[nt]);
    }
    __syncthreads();
  }

#pragma unroll
  for (int e = 0; e < 8; ++e) {
    float inv = 1.0f / lrun[e];
    int s = q0 + wv * 16 + hi * 8 + e;
    size_t base = ((size_t)b * SS + s) * HID + h * HD;
#pragma unroll
    for (int nt = 0; nt < 4; ++nt)
      Ctx[base + nt * 16 + lrow] = (bf16)(accc[nt][e] * inv);
  }
}

// ---------------------------------------------------------------------------
// Host-side orchestration
// ---------------------------------------------------------------------------
extern "C" void kernel_launch(void* const* d_in, const int* in_sizes, int n_in,
                              void* d_out, int out_size, void* d_ws, size_t ws_size,
                              hipStream_t stream) {
  const float* hidden = (const float*)d_in[0];
  const float* amask  = (const float*)d_in[1];
  const float* Wq = (const float*)d_in[2];  const float* bq = (const float*)d_in[3];
  const float* Wk = (const float*)d_in[4];  const float* bk = (const float*)d_in[5];
  const float* Wv = (const float*)d_in[6];  const float* bv = (const float*)d_in[7];
  const float* Wo = (const float*)d_in[8];  const float* bo = (const float*)d_in[9];
  const float* rel = (const float*)d_in[10];
  float* out = (float*)d_out;

  char* ws = (char*)d_ws;
  constexpr size_t XB_N = (size_t)BB * SS * HID;
  constexpr size_t W_N  = (size_t)HID * HID;
  constexpr size_t XB_O = 0;
  constexpr size_t WQ_O = XB_O + XB_N * sizeof(bf16);
  constexpr size_t WK_O = WQ_O + W_N * sizeof(bf16);
  constexpr size_t WV_O = WK_O + W_N * sizeof(bf16);
  constexpr size_t WO_O = WV_O + W_N * sizeof(bf16);
  constexpr size_t QH_O = WO_O + W_N * sizeof(bf16);
  constexpr size_t KH_O = QH_O + XB_N * sizeof(bf16);
  constexpr size_t VT_O = KH_O + XB_N * sizeof(bf16);
  constexpr size_t CT_O = VT_O + XB_N * sizeof(bf16);

  bf16* Xb  = (bf16*)(ws + XB_O);
  bf16* Wqt = (bf16*)(ws + WQ_O);
  bf16* Wkt = (bf16*)(ws + WK_O);
  bf16* Wvt = (bf16*)(ws + WV_O);
  bf16* Wot = (bf16*)(ws + WO_O);
  bf16* Qh  = (bf16*)(ws + QH_O);
  bf16* Kh  = (bf16*)(ws + KH_O);
  bf16* Vt  = (bf16*)(ws + VT_O);
  bf16* Ctx = (bf16*)(ws + CT_O);

  // 1) Conversions: hidden -> bf16; weights -> bf16 transposed [n][k].
  cvt_bf16_kernel<<<(int)((XB_N + 255) / 256), 256, 0, stream>>>(hidden, Xb, (int)XB_N);
  dim3 tgrid(16, 16);
  cvt_bf16_transpose_kernel<<<tgrid, 256, 0, stream>>>(Wq, Wqt);
  cvt_bf16_transpose_kernel<<<tgrid, 256, 0, stream>>>(Wk, Wkt);
  cvt_bf16_transpose_kernel<<<tgrid, 256, 0, stream>>>(Wv, Wvt);
  cvt_bf16_transpose_kernel<<<tgrid, 256, 0, stream>>>(Wo, Wot);

  // 2) QKV projections (Q folded with 1/sqrt(HD); V written transposed).
  dim3 ggrid(64, 8);
  gemm512<1><<<ggrid, 256, 0, stream>>>(Xb, Wqt, bq, Qh, 0.125f);
  gemm512<1><<<ggrid, 256, 0, stream>>>(Xb, Wkt, bk, Kh, 1.0f);
  gemm512<2><<<ggrid, 256, 0, stream>>>(Xb, Wvt, bv, Vt, 1.0f);

  // 3) Flash attention.
  dim3 agrid(SS / 128, BB * NH);
  attn_kernel<<<agrid, 256, 0, stream>>>(Qh, Kh, Vt, rel, amask, Ctx);

  // 4) Output projection -> f32 d_out.
  gemm512<0><<<ggrid, 256, 0, stream>>>(Ctx, Wot, bo, out, 1.0f);
}
